// ExaoneSelfAttention_82154134438141
// MI455X (gfx1250) — matrix operand order repaired and checked by reading, not compiled, not run on hardware
//
#include <hip/hip_runtime.h>
#include <hip/hip_bf16.h>

// ---------------- problem constants ----------------
#define BB   2
#define SS   2048
#define DD   2048
#define HH   16
#define HDD  128
#define MTOT (BB*SS)           // 4096 rows
#define CAPV 50.0f
#define FMIN_ (-3.402823466e38f)

typedef __bf16 bf16_t;
typedef __attribute__((ext_vector_type(16))) __bf16 v16bf;
typedef __attribute__((ext_vector_type(8)))  __bf16 v8bf;
typedef __attribute__((ext_vector_type(8)))  float  v8f;

__device__ __forceinline__ v16bf load_v16(const bf16_t* p0, const bf16_t* p1) {
    union U { v16bf v; v8bf h[2]; } u;
    u.h[0] = *(const v8bf*)p0;
    u.h[1] = *(const v8bf*)p1;
    return u.v;
}

__device__ __forceinline__ v8f wmma_bf16(v16bf a, v16bf b, v8f c) {
    // D = A(16x32 bf16) * B(32x16 bf16) + C(16x16 f32)
    return __builtin_amdgcn_wmma_f32_16x16x32_bf16(
        /*neg_a=*/false, a, /*neg_b=*/false, b,
        /*c_mod=*/(short)0, c, /*reuse_a=*/false, /*reuse_b=*/false);
}

__device__ __forceinline__ float captanh(float x) {
    float t = x * (1.0f / CAPV);
    t = fminf(fmaxf(t, -15.0f), 15.0f);
    float e = __expf(2.0f * t);
    return CAPV * ((e - 1.0f) / (e + 1.0f));
}

// -------- CDNA5 async global->LDS copy (ASYNCcnt path), with safe fallback --------
#if defined(__HIP_DEVICE_COMPILE__) && __has_builtin(__builtin_amdgcn_global_load_async_to_lds_b128)
#define ASYNC_LDS 1
typedef __attribute__((__vector_size__(4 * sizeof(int)))) int gv4i;
typedef __attribute__((address_space(1))) gv4i* gptr_t;
typedef __attribute__((address_space(3))) gv4i* lptr_t;
__device__ __forceinline__ void async_cp16(const bf16_t* g, bf16_t* l) {
    __builtin_amdgcn_global_load_async_to_lds_b128(
        (gptr_t)(void*)g, (lptr_t)(void*)l, 0, 0);
}
__device__ __forceinline__ void async_wait0() {
#if __has_builtin(__builtin_amdgcn_s_wait_asynccnt)
    __builtin_amdgcn_s_wait_asynccnt(0);
#else
    asm volatile("s_wait_asynccnt 0x0" ::: "memory");
#endif
}
#endif

// ---------------- elementwise converts ----------------
__global__ void f2bf_kernel(const float* __restrict__ s, bf16_t* __restrict__ d, int n) {
    int i = blockIdx.x * blockDim.x + threadIdx.x;
    if (i < n) d[i] = (bf16_t)s[i];
}

// w is [K,N] row-major (K=N=D); wT output [N,K] row-major bf16
__global__ void transpose_f2bf_kernel(const float* __restrict__ w, bf16_t* __restrict__ wT) {
    int i = blockIdx.x * blockDim.x + threadIdx.x;
    if (i >= DD * DD) return;
    int n = i >> 11;           // / 2048
    int k = i & (DD - 1);
    wT[i] = (bf16_t)w[(size_t)k * DD + n];
}

// ---------------- generic bf16 GEMM: C(f32)[M,N] = A[M,K] * Bt[N,K]^T ----------------
#define KSTEP   64
#define BSTRIDE 72              // LDS row stride in elements (144B, 16B-aligned, pads banks)

__global__ __launch_bounds__(256) void gemm_bf16_kernel(
    const bf16_t* __restrict__ A, const bf16_t* __restrict__ Bt,
    float* __restrict__ C, int Mtot, int Ntot, int Ktot) {
    __shared__ bf16_t ldsB[2 * 64 * BSTRIDE];   // double-buffered 64n x 64k bf16 tiles

    const int tid  = threadIdx.x;
    const int lane = tid & 31;
    const int wave = tid >> 5;
    const int h_   = lane >> 4;
    const int lrow = lane & 15;
    const int m0   = blockIdx.y * 128 + wave * 16;
    const int n0   = blockIdx.x * 64;

    v8f acc[4];
#pragma unroll
    for (int i = 0; i < 4; ++i) acc[i] = (v8f){0,0,0,0,0,0,0,0};

    const bf16_t* arow = A + (size_t)(m0 + lrow) * Ktot;   // A operand: row M=lane&15

    // cooperative staging: 256 threads x 32B = 8KB tile (64 rows x 64 k-elems)
    const int srow = tid >> 2;            // 0..63
    const int scol = (tid & 3) * 16;      // 0,16,32,48 elements
    const bf16_t* bsrc  = Bt + (size_t)(n0 + srow) * Ktot + scol;
    bf16_t*       ldst0 = &ldsB[srow * BSTRIDE + scol];
    bf16_t*       ldst1 = ldst0 + 64 * BSTRIDE;

#if defined(ASYNC_LDS)
    // prologue: kick off tile 0 into buffer 0; preload A operands for k0=0
    async_cp16(bsrc, ldst0);
    async_cp16(bsrc + 8, ldst0 + 8);
    v16bf a0 = load_v16(arow + 8 * h_, arow + 16 + 8 * h_);
    v16bf a1 = load_v16(arow + 32 + 8 * h_, arow + 48 + 8 * h_);
    int buf = 0;
    for (int k0 = 0; k0 < Ktot; k0 += KSTEP) {
        async_wait0();          // this wave's contributions to tile k0 are in LDS
        __syncthreads();        // all waves staged tile k0; all done reading other buffer
        if (k0 + KSTEP < Ktot) {  // issue next tile into the other buffer (overlaps WMMA)
            bf16_t* nd = buf ? ldst0 : ldst1;
            async_cp16(bsrc + k0 + KSTEP, nd);
            async_cp16(bsrc + k0 + KSTEP + 8, nd + 8);
        }
        // batch all 8 B operands from LDS into registers
        const bf16_t* lb = &ldsB[buf * 64 * BSTRIDE];
        v16bf bo[8];
#pragma unroll
        for (int kc = 0; kc < 2; ++kc)
#pragma unroll
            for (int nt = 0; nt < 4; ++nt) {
                const bf16_t* bp = &lb[(nt * 16 + lrow) * BSTRIDE + kc * 32 + 8 * h_];
                bo[kc * 4 + nt] = load_v16(bp, bp + 16);
            }
        // software-pipeline the next A operands (wait lands next iteration)
        v16bf na0 = a0, na1 = a1;
        if (k0 + KSTEP < Ktot) {
            const bf16_t* an = arow + k0 + KSTEP;
            na0 = load_v16(an + 8 * h_, an + 16 + 8 * h_);
            na1 = load_v16(an + 32 + 8 * h_, an + 48 + 8 * h_);
        }
        // 8 back-to-back WMMAs
#pragma unroll
        for (int nt = 0; nt < 4; ++nt) acc[nt] = wmma_bf16(a0, bo[nt], acc[nt]);
#pragma unroll
        for (int nt = 0; nt < 4; ++nt) acc[nt] = wmma_bf16(a1, bo[4 + nt], acc[nt]);
        a0 = na0;
        a1 = na1;
        buf ^= 1;
    }
#else
    // fallback: register double-buffer through VGPRs (overlap global latency with WMMA)
    v8bf r0 = *(const v8bf*)bsrc;
    v8bf r1 = *(const v8bf*)(bsrc + 8);
    for (int k0 = 0; k0 < Ktot; k0 += KSTEP) {
        __syncthreads();        // everyone done reading previous tile
        *(v8bf*)ldst0       = r0;
        *(v8bf*)(ldst0 + 8) = r1;
        if (k0 + KSTEP < Ktot) {
            r0 = *(const v8bf*)(bsrc + k0 + KSTEP);
            r1 = *(const v8bf*)(bsrc + k0 + KSTEP + 8);
        }
        __syncthreads();        // tile k0 visible in LDS
#pragma unroll
        for (int kc = 0; kc < 2; ++kc) {
            v16bf a = load_v16(arow + k0 + kc * 32 + 8 * h_,
                               arow + k0 + kc * 32 + 16 + 8 * h_);
#pragma unroll
            for (int nt = 0; nt < 4; ++nt) {
                const bf16_t* bp = &ldsB[(nt * 16 + lrow) * BSTRIDE + kc * 32 + 8 * h_];
                acc[nt] = wmma_bf16(a, load_v16(bp, bp + 16), acc[nt]);
            }
        }
    }
#endif

#pragma unroll
    for (int nt = 0; nt < 4; ++nt)
#pragma unroll
        for (int r = 0; r < 8; ++r)
            C[(size_t)(m0 + r + 8 * h_) * Ntot + n0 + nt * 16 + lrow] = acc[nt][r];
}

// ---------------- RoPE + relayout [B,S,D]f32 -> [B,H,S,HD]bf16 ----------------
__global__ void rope_relayout_kernel(const float* __restrict__ src, bf16_t* __restrict__ dst) {
    int i = blockIdx.x * blockDim.x + threadIdx.x;
    if (i >= BB * SS * HH * (HDD / 2)) return;
    int j  = i & 63;
    int hh = (i >> 6) & (HH - 1);
    int s  = (i >> 10) & (SS - 1);
    int b  = i >> 21;
    // inv_freq = 10000^(-2j/128)
    float inv = __expf(-(2.0f * (float)j / (float)HDD) * 9.210340372f);
    float ang = (float)s * inv;
    float sn, cs;
    __sincosf(ang, &sn, &cs);
    size_t sb = ((size_t)b * SS + s) * DD + (size_t)hh * HDD;
    float x1 = src[sb + j], x2 = src[sb + 64 + j];
    size_t db = (((size_t)b * HH + hh) * SS + s) * (size_t)HDD;
    dst[db + j]      = (bf16_t)(x1 * cs - x2 * sn);
    dst[db + 64 + j] = (bf16_t)(x2 * cs + x1 * sn);
}

// ---------------- V relayout [B,S,D]f32 -> Vt [B,H,HD,S]bf16 ----------------
__global__ void v_relayout_kernel(const float* __restrict__ src, bf16_t* __restrict__ Vt) {
    int i = blockIdx.x * blockDim.x + threadIdx.x;
    if (i >= BB * SS * DD) return;
    int d  = i & (HDD - 1);
    int hh = (i >> 7) & (HH - 1);
    int s  = (i >> 11) & (SS - 1);
    int b  = i >> 22;
    Vt[(((size_t)b * HH + hh) * HDD + d) * SS + s] =
        (bf16_t)src[((size_t)b * SS + s) * DD + (size_t)hh * HDD + d];
}

// ---------------- flash attention (lingvo softmax + tanh cap, causal) ----------------
// Qb,Kb: [B,H,S,HD] bf16 (rotary applied); Vt: [B,H,HD,S] bf16; ctx out: [B,S,D] bf16
__global__ __launch_bounds__(128) void attn_kernel(
    const bf16_t* __restrict__ Qb, const bf16_t* __restrict__ Kb,
    const bf16_t* __restrict__ Vt, bf16_t* __restrict__ ctx) {
    __shared__ bf16_t ldsP[4 * 16 * 40];   // per-wave 16x32 prob tile (stride 40)

    const int tid  = threadIdx.x;
    const int lane = tid & 31;
    const int wave = tid >> 5;
    const int h_   = lane >> 4;
    const int lrow = lane & 15;
    const int bh   = blockIdx.y;
    const int b    = bh / HH;
    const int hh   = bh % HH;
    const int qbase = (blockIdx.x * 4 + wave) * 16;
    bf16_t* lp = &ldsP[wave * 16 * 40];

    // resident Q A-operands for all 4 hd-chunks of 32
    v16bf qa[4];
    const bf16_t* qrow = Qb + ((size_t)bh * SS + qbase + lrow) * (size_t)HDD;
#pragma unroll
    for (int c = 0; c < 4; ++c)
        qa[c] = load_v16(qrow + c * 32 + 8 * h_, qrow + c * 32 + 16 + 8 * h_);

    v8f acc[8];
    float m[8], lsum[8];
#pragma unroll
    for (int n = 0; n < 8; ++n) acc[n] = (v8f){0,0,0,0,0,0,0,0};
#pragma unroll
    for (int r = 0; r < 8; ++r) { m[r] = 0.0f; lsum[r] = 1.0f; }  // phantom exp(-m) term

    const int nkb = (qbase + 16 + 31) >> 5;   // causal: only key blocks with kb <= q_max
    for (int t = 0; t < nkb; ++t) {
        const int kb = t * 32;
        v8f s0 = (v8f){0,0,0,0,0,0,0,0};
        v8f s1 = (v8f){0,0,0,0,0,0,0,0};
#pragma unroll
        for (int c = 0; c < 4; ++c) {
            const bf16_t* kr = Kb + ((size_t)bh * SS + kb + lrow) * (size_t)HDD + c * 32 + 8 * h_;
            s0 = wmma_bf16(qa[c], load_v16(kr, kr + 16), s0);
            const bf16_t* kr1 = kr + (size_t)16 * HDD;
            s1 = wmma_bf16(qa[c], load_v16(kr1, kr1 + 16), s1);
        }
        float alpha[8];
#pragma unroll
        for (int r = 0; r < 8; ++r) {
            const int q = qbase + r + 8 * h_;
            float v0 = (kb + lrow      <= q) ? s0[r] : FMIN_;
            float v1 = (kb + 16 + lrow <= q) ? s1[r] : FMIN_;
            v0 = captanh(v0);
            v1 = captanh(v1);
            float mx = fmaxf(v0, v1);
            mx = fmaxf(mx, __shfl_xor(mx, 1, 32));
            mx = fmaxf(mx, __shfl_xor(mx, 2, 32));
            mx = fmaxf(mx, __shfl_xor(mx, 4, 32));
            mx = fmaxf(mx, __shfl_xor(mx, 8, 32));
            float mn = fmaxf(m[r], mx);
            float al = __expf(m[r] - mn);
            float p0 = __expf(v0 - mn);
            float p1 = __expf(v1 - mn);
            float ps = p0 + p1;
            ps += __shfl_xor(ps, 1, 32);
            ps += __shfl_xor(ps, 2, 32);
            ps += __shfl_xor(ps, 4, 32);
            ps += __shfl_xor(ps, 8, 32);
            lsum[r] = lsum[r] * al + ps;
            m[r] = mn;
            alpha[r] = al;
            lp[(r + 8 * h_) * 40 + lrow]      = (bf16_t)p0;
            lp[(r + 8 * h_) * 40 + 16 + lrow] = (bf16_t)p1;
        }
#pragma unroll
        for (int n = 0; n < 8; ++n)
#pragma unroll
            for (int r = 0; r < 8; ++r) acc[n][r] *= alpha[r];

        // cross-lane LDS RAW inside the wave: ensure the ds_store_b16's completed
        asm volatile("s_wait_dscnt 0" ::: "memory");

        const bf16_t* pr = lp + lrow * 40;
        v16bf pa = load_v16(pr + 8 * h_, pr + 16 + 8 * h_);
#pragma unroll
        for (int n = 0; n < 8; ++n) {
            const bf16_t* vr = Vt + ((size_t)bh * HDD + n * 16 + lrow) * (size_t)SS + kb + 8 * h_;
            acc[n] = wmma_bf16(pa, load_v16(vr, vr + 16), acc[n]);
        }
    }

    // epilogue: normalize and write ctx [B,S,D] bf16
#pragma unroll
    for (int n = 0; n < 8; ++n)
#pragma unroll
        for (int r = 0; r < 8; ++r) {
            float v = acc[n][r] / lsum[r];
            size_t row = (size_t)b * SS + qbase + r + 8 * h_;
            ctx[row * DD + (size_t)hh * HDD + n * 16 + lrow] = (bf16_t)v;
        }
}

// ---------------- host-side launcher ----------------
extern "C" void kernel_launch(void* const* d_in, const int* in_sizes, int n_in,
                              void* d_out, int out_size, void* d_ws, size_t ws_size,
                              hipStream_t stream) {
    (void)in_sizes; (void)n_in; (void)out_size; (void)ws_size;
    const float* hs = (const float*)d_in[0];
    const float* wq = (const float*)d_in[1];
    const float* wk = (const float*)d_in[2];
    const float* wv = (const float*)d_in[3];
    const float* wo = (const float*)d_in[4];

    // workspace carve-up (~128 MB total, intermediates live in the 192MB L2)
    char* w = (char*)d_ws;
    bf16_t* hb  = (bf16_t*)w; w += (size_t)MTOT * DD * 2;           // 16 MB
    bf16_t* wqT = (bf16_t*)w; w += (size_t)DD * DD * 2;             //  8 MB
    bf16_t* wkT = (bf16_t*)w; w += (size_t)DD * DD * 2;
    bf16_t* wvT = (bf16_t*)w; w += (size_t)DD * DD * 2;
    bf16_t* woT = (bf16_t*)w; w += (size_t)DD * DD * 2;
    bf16_t* Qb  = (bf16_t*)w; w += (size_t)BB * HH * SS * HDD * 2;  // 16 MB
    bf16_t* Kb  = (bf16_t*)w; w += (size_t)BB * HH * SS * HDD * 2;
    bf16_t* Vt  = (bf16_t*)w; w += (size_t)BB * HH * SS * HDD * 2;
    float*  tmp = (float*)w;                                        // 32 MB f32 scratch
    bf16_t* ctx = (bf16_t*)tmp;                                     // reuse after post-V

    const int nh = MTOT * DD;
    f2bf_kernel<<<(nh + 255) / 256, 256, 0, stream>>>(hs, hb, nh);
    const int nw = DD * DD;
    transpose_f2bf_kernel<<<(nw + 255) / 256, 256, 0, stream>>>(wq, wqT);
    transpose_f2bf_kernel<<<(nw + 255) / 256, 256, 0, stream>>>(wk, wkT);
    transpose_f2bf_kernel<<<(nw + 255) / 256, 256, 0, stream>>>(wv, wvT);
    transpose_f2bf_kernel<<<(nw + 255) / 256, 256, 0, stream>>>(wo, woT);

    dim3 gg(DD / 64, MTOT / 128);
    const int nr = BB * SS * HH * (HDD / 2);
    const int nv = BB * SS * DD;

    gemm_bf16_kernel<<<gg, 256, 0, stream>>>(hb, wqT, tmp, MTOT, DD, DD);
    rope_relayout_kernel<<<(nr + 255) / 256, 256, 0, stream>>>(tmp, Qb);

    gemm_bf16_kernel<<<gg, 256, 0, stream>>>(hb, wkT, tmp, MTOT, DD, DD);
    rope_relayout_kernel<<<(nr + 255) / 256, 256, 0, stream>>>(tmp, Kb);

    gemm_bf16_kernel<<<gg, 256, 0, stream>>>(hb, wvT, tmp, MTOT, DD, DD);
    v_relayout_kernel<<<(nv + 255) / 256, 256, 0, stream>>>(tmp, Vt);

    attn_kernel<<<dim3(SS / 64, BB * HH), 128, 0, stream>>>(Qb, Kb, Vt, ctx);

    gemm_bf16_kernel<<<gg, 256, 0, stream>>>(ctx, woT, (float*)d_out, MTOT, DD, DD);
}
